// CKConv1D_24129126269468
// MI455X (gfx1250) — matrix operand order repaired and checked by reading, not compiled, not run on hardware
//
#include <hip/hip_runtime.h>
#include <hip/hip_bf16.h>

typedef __attribute__((ext_vector_type(16))) _Float16 v16h;
typedef __attribute__((ext_vector_type(8)))  float    v8f;

#define L_SEQ 256
#define CIN   8
#define COUT  8
#define HDIM  32

// ---------------------------------------------------------------------------
// ws layout (bytes):
//   [0,    384)  W1f32 [32][3]      weight-normed SIREN layer-1
//   [512,  2560) W2f16 [32][32]     weight-normed layer-2, row k = W2[k][h]
//   [2560, 6656) xf16  [256][8]     f16 copy of x
//   [6656,39424) Kf16  [2048][8]    K[(c*256+d)][o]  (conv B-matrix layout)
// ---------------------------------------------------------------------------

__global__ void ck_prep(const float* __restrict__ x,
                        const float* __restrict__ v1, const float* __restrict__ g1,
                        const float* __restrict__ v2, const float* __restrict__ g2,
                        float* __restrict__ W1, _Float16* __restrict__ W2h,
                        _Float16* __restrict__ xh) {
  const int t = threadIdx.x;
  if (t < HDIM) {
    float n1 = 0.0f;
#pragma unroll
    for (int j = 0; j < 3; ++j) { float v = v1[t * 3 + j]; n1 += v * v; }
    const float s1 = g1[t] * rsqrtf(n1);
#pragma unroll
    for (int j = 0; j < 3; ++j) W1[t * 3 + j] = v1[t * 3 + j] * s1;

    float n2 = 0.0f;
#pragma unroll
    for (int h = 0; h < HDIM; ++h) { float v = v2[t * HDIM + h]; n2 += v * v; }
    const float s2 = g2[t] * rsqrtf(n2);
#pragma unroll
    for (int h = 0; h < HDIM; ++h)
      W2h[t * HDIM + h] = (_Float16)(v2[t * HDIM + h] * s2);
  }
  for (int idx = t; idx < L_SEQ * CIN; idx += blockDim.x)
    xh[idx] = (_Float16)x[idx];
}

// One wave per 16 (o,c,d) points. A = sin(layer1) 16x32 f16, B = W2^T tiles.
__global__ void __launch_bounds__(32)
ck_kernelnet(const float* __restrict__ W1, const float* __restrict__ b1,
             const _Float16* __restrict__ W2h, const float* __restrict__ b2,
             const float* __restrict__ w3, const float* __restrict__ b3,
             _Float16* __restrict__ Kh) {
  const int lane = threadIdx.x & 31;
  const int m    = lane & 15;      // A row (point in tile) AND B/D column n
  const int half = lane >> 4;
  const int T    = blockIdx.x;     // 0..1023
  const int oc   = T >> 4;
  const int dbase = (T & 15) << 4;
  const int o    = oc >> 3;
  const int c    = oc & 7;

  // Layer 1: this lane's A-matrix slice (row m, K-subset per ISA f16 A layout)
  const float dt = -(float)(dbase + m) * (1.0f / 256.0f);
  const float fc = (float)c, fo = (float)o;
  v16h a;
#pragma unroll
  for (int e = 0; e < 16; ++e) {
    const int h = (e & 7) + (half << 3) + ((e >> 3) << 4);
    const float z = dt * W1[h * 3 + 0] + fc * W1[h * 3 + 1] +
                    fo * W1[h * 3 + 2] + b1[h];
    a[e] = (_Float16)__sinf(z);
  }

  // B tiles: B[K=h][N=k]; lane column n=m, rows h in same K layout as A
  const int h0 = half << 4;
  v16h bt0, bt1;
#pragma unroll
  for (int e = 0; e < 16; ++e) {
    bt0[e] = W2h[m * HDIM + (h0 >> 1) + (e & 7) + ((e >> 3) << 4)];          // k = n
    bt1[e] = W2h[(m + 16) * HDIM + (h0 >> 1) + (e & 7) + ((e >> 3) << 4)];   // k = n + 16
  }

  v8f cz = {};
  v8f acc0 = __builtin_amdgcn_wmma_f32_16x16x32_f16(false, a, false, bt0,
                                                    (short)0, cz, false, false);
  v8f acc1 = __builtin_amdgcn_wmma_f32_16x16x32_f16(false, a, false, bt1,
                                                    (short)0, cz, false, false);

  // sin + dot with w3 (reduce across the 16 D columns per half-wave)
  const float b2a = b2[m], b2b = b2[m + 16];
  const float w3a = w3[m], w3b = w3[m + 16];
  const float bias3 = b3[0];
#pragma unroll
  for (int r = 0; r < 8; ++r) {
    float val = __sinf(acc0[r] + b2a) * w3a + __sinf(acc1[r] + b2b) * w3b;
    val += __shfl_xor(val, 1, 32);
    val += __shfl_xor(val, 2, 32);
    val += __shfl_xor(val, 4, 32);
    val += __shfl_xor(val, 8, 32);
    if (m == 0) {
      const int pt = r + (half << 3);         // point index (D row)
      const int d  = dbase + pt;
      Kh[(c * 256 + d) * COUT + o] = (_Float16)(val + bias3);
    }
  }
}

// Causal Toeplitz conv as GEMM: out[16 x 8] tile, K-dim = (c,d) of length 2048.
// A[i][kk] = x[i-d][c] with src<0 -> 0 (this IS the causal mask).
__global__ void __launch_bounds__(32)
ck_conv(const _Float16* __restrict__ xh, const _Float16* __restrict__ Kh,
        float* __restrict__ out) {
  const int lane = threadIdx.x & 31;
  const int m    = lane & 15;      // A row within tile AND B/D column n (= o)
  const int half = lane >> 4;
  const int I0   = blockIdx.x << 4;

  v8f acc = {};
  for (int ch = 0; ch < 64; ++ch) {
    const int c  = ch >> 3;
    const int db = (ch & 7) << 5;
    v16h a, b;
#pragma unroll
    for (int e = 0; e < 16; ++e) {
      const int kk  = (e & 7) + (half << 3) + ((e >> 3) << 4);  // A K layout
      const int src = I0 + m - (db + kk);
      a[e] = (src >= 0) ? xh[src * CIN + c] : (_Float16)0.0f;
      const int kk2 = (e & 7) + (half << 3) + ((e >> 3) << 4);   // B K layout
      b[e] = (m < COUT) ? Kh[(c * 256 + db + kk2) * COUT + m] : (_Float16)0.0f;
    }
    acc = __builtin_amdgcn_wmma_f32_16x16x32_f16(false, a, false, b,
                                                 (short)0, acc, false, false);
  }
  if (m < COUT) {
#pragma unroll
    for (int r = 0; r < 8; ++r) {
      const int row = I0 + r + (half << 3);
      out[row * COUT + m] = acc[r];
    }
  }
}

extern "C" void kernel_launch(void* const* d_in, const int* in_sizes, int n_in,
                              void* d_out, int out_size, void* d_ws, size_t ws_size,
                              hipStream_t stream) {
  const float* x  = (const float*)d_in[0];
  // d_in[1] = t (uniform grid; dt derived analytically)
  const float* v1 = (const float*)d_in[2];
  const float* g1 = (const float*)d_in[3];
  const float* b1 = (const float*)d_in[4];
  const float* v2 = (const float*)d_in[5];
  const float* g2 = (const float*)d_in[6];
  const float* b2 = (const float*)d_in[7];
  const float* w3 = (const float*)d_in[8];
  const float* b3 = (const float*)d_in[9];
  float* out = (float*)d_out;

  char* ws = (char*)d_ws;
  float*    W1f = (float*)(ws + 0);
  _Float16* W2h = (_Float16*)(ws + 512);
  _Float16* xh  = (_Float16*)(ws + 512 + 2048);
  _Float16* Kh  = (_Float16*)(ws + 512 + 2048 + 4096);

  ck_prep<<<1, 256, 0, stream>>>(x, v1, g1, v2, g2, W1f, W2h, xh);
  ck_kernelnet<<<1024, 32, 0, stream>>>(W1f, b1, W2h, b2, w3, b3, Kh);
  ck_conv<<<16, 32, 0, stream>>>(xh, Kh, out);
}
